// PointnetFPModule_39539468927437
// MI455X (gfx1250) — compile-verified, matrix operand order checked by reading, not run on hardware
//
#include <hip/hip_runtime.h>
#include <hip/hip_bf16.h>
#include <cstdint>

// ---------------------------------------------------------------------------
// Problem constants (match reference)
// ---------------------------------------------------------------------------
#define PB    16          // batch
#define PN    4096        // unknown points
#define PM    1024        // known points
#define C1C   256         // unknow_feats channels
#define C2C   512         // known_feats channels
#define CIN   768         // C1 + C2
#define HC    256         // hidden channels
#define SEC   10          // SE bottleneck
#define NCOL  (PB * PN)   // 65536 GEMM columns
#define BN_EPS 1e-5f

typedef __attribute__((ext_vector_type(16))) __bf16 v16bf;
typedef __attribute__((ext_vector_type(8)))  float  v8f;

union FragU {               // raw bits view of a 16x32 bf16 WMMA operand (8 VGPRs)
    uint4 u[2];
    v16bf v;
};

static __device__ __forceinline__ unsigned short f32_to_bf16(float f) {
    unsigned int u = __float_as_uint(f);
    u += 0x7FFFu + ((u >> 16) & 1u);   // round-to-nearest-even
    return (unsigned short)(u >> 16);
}
static __device__ __forceinline__ float dev_sigmoid(float x) {
    return 1.0f / (1.0f + __expf(-x));
}

// ---------------------------------------------------------------------------
// K1: three_nn — per point top-3 nearest known points + inverse-distance weights
// grid (PN/128, PB), block 128
// ---------------------------------------------------------------------------
__global__ __launch_bounds__(128) void k_three_nn(
        const float* __restrict__ unknown,   // [B,N,3]
        const float* __restrict__ known,     // [B,M,3]
        int*   __restrict__ idx3,            // [B*N,3]
        float* __restrict__ w3)              // [B*N,3]
{
    __shared__ float kx[PM], ky[PM], kz[PM];
    const int b = blockIdx.y;
    const float* kb = known + (size_t)b * PM * 3;
    for (int m = threadIdx.x; m < PM; m += 128) {
        kx[m] = kb[m * 3 + 0];
        ky[m] = kb[m * 3 + 1];
        kz[m] = kb[m * 3 + 2];
    }
    __syncthreads();

    const int n = blockIdx.x * 128 + threadIdx.x;
    const float* up = unknown + ((size_t)b * PN + n) * 3;
    const float ux = up[0], uy = up[1], uz = up[2];

    float b0 = 3.4e38f, b1 = 3.4e38f, b2 = 3.4e38f;
    int   i0 = 0, i1 = 0, i2 = 0;
    for (int m = 0; m < PM; ++m) {
        const float dx = ux - kx[m], dy = uy - ky[m], dz = uz - kz[m];
        const float d = dx * dx + dy * dy + dz * dz;
        if (d < b2) {
            if (d < b1) {
                if (d < b0) { b2 = b1; i2 = i1; b1 = b0; i1 = i0; b0 = d; i0 = m; }
                else        { b2 = b1; i2 = i1; b1 = d;  i1 = m; }
            } else          { b2 = d;  i2 = m; }
        }
    }
    const float r0 = 1.0f / (b0 + 1e-8f);
    const float r1 = 1.0f / (b1 + 1e-8f);
    const float r2 = 1.0f / (b2 + 1e-8f);
    const float rs = 1.0f / (r0 + r1 + r2);

    const size_t col = (size_t)b * PN + n;
    idx3[col * 3 + 0] = i0;  w3[col * 3 + 0] = r0 * rs;
    idx3[col * 3 + 1] = i1;  w3[col * 3 + 1] = r1 * rs;
    idx3[col * 3 + 2] = i2;  w3[col * 3 + 2] = r2 * rs;
}

// ---------------------------------------------------------------------------
// K2: interpolate + concat -> X (bf16, column-major: X[col*CIN + c])
// grid NCOL/8, block 256 (8 columns per block)
// ---------------------------------------------------------------------------
__global__ __launch_bounds__(256) void k_interp_pack(
        const float* __restrict__ uf,        // [B,C1,N]
        const float* __restrict__ kf,        // [B,C2,M]
        const int*   __restrict__ idx3,
        const float* __restrict__ w3,
        unsigned short* __restrict__ X)      // [NCOL][CIN] bf16
{
    const size_t colbase = (size_t)blockIdx.x * 8;
    for (int cc = 0; cc < 8; ++cc) {
        const size_t col = colbase + cc;
        const int b = (int)(col >> 12);
        const int n = (int)(col & 4095);
        const int j0 = idx3[col * 3 + 0];
        const int j1 = idx3[col * 3 + 1];
        const int j2 = idx3[col * 3 + 2];
        const float w0 = w3[col * 3 + 0];
        const float w1 = w3[col * 3 + 1];
        const float w2 = w3[col * 3 + 2];
        for (int c = threadIdx.x; c < CIN; c += 256) {
            float v;
            if (c < C2C) {
                const float* base = kf + ((size_t)(b * C2C + c)) * PM;
                v = w0 * base[j0] + w1 * base[j1] + w2 * base[j2];
            } else {
                v = uf[((size_t)(b * C1C + (c - C2C))) * PN + n];
            }
            X[col * CIN + c] = f32_to_bf16(v);
        }
    }
}

// ---------------------------------------------------------------------------
// K3: pack W1 [HC x CIN] and W2 [HC x HC] to bf16 row-major
// grid 768, block 256
// ---------------------------------------------------------------------------
__global__ __launch_bounds__(256) void k_pack_weights(
        const float* __restrict__ W1, const float* __restrict__ W2,
        unsigned short* __restrict__ Wb1, unsigned short* __restrict__ Wb2)
{
    const int t = blockIdx.x * 256 + threadIdx.x;
    if (t < HC * CIN) Wb1[t] = f32_to_bf16(W1[t]);
    if (t < HC * HC)  Wb2[t] = f32_to_bf16(W2[t]);
}

// ---------------------------------------------------------------------------
// K4/K7: bf16 WMMA GEMM.  D[256 x NCOL] (f32 row-major) = A[256 x K] * B[K x NCOL]
//   A: bf16 row-major (K contiguous)   B: bf16 column-major (K contiguous per col)
//   Block = 8 waves; wave tile 64(M) x 32(N); block tile 256 x 64; K step 32.
//   Per lane, each 16x32 bf16 fragment = two b128 loads (K = 8h..8h+7 / 16+8h..).
// grid NCOL/64, block 256
// ---------------------------------------------------------------------------
__global__ __launch_bounds__(256) void k_gemm_bf16(
        const unsigned short* __restrict__ A,
        const unsigned short* __restrict__ Bm,
        float* __restrict__ D, int K)
{
    const int lane = threadIdx.x & 31;
    const int wid  = threadIdx.x >> 5;
    const int m16  = lane & 15;
    const int kh   = lane >> 4;                  // lane half: selects K group
    const int row0 = (wid & 3) * 64;             // 4 waves along M
    const int col0 = blockIdx.x * 64 + (wid >> 2) * 32;  // 2 waves along N

    const unsigned short* Arow[4];
    const unsigned short* Bcol[2];
#pragma unroll
    for (int i = 0; i < 4; ++i) Arow[i] = A  + (size_t)(row0 + i * 16 + m16) * K;
#pragma unroll
    for (int j = 0; j < 2; ++j) Bcol[j] = Bm + (size_t)(col0 + j * 16 + m16) * K;

    const v8f vzero = {0.f, 0.f, 0.f, 0.f, 0.f, 0.f, 0.f, 0.f};
    v8f acc[4][2];
#pragma unroll
    for (int i = 0; i < 4; ++i)
#pragma unroll
        for (int j = 0; j < 2; ++j) acc[i][j] = vzero;

    const int koff = 8 * kh;
    for (int k0 = 0; k0 < K; k0 += 32) {
        FragU af[4], bf[2];
#pragma unroll
        for (int i = 0; i < 4; ++i) {
            const uint4* p = reinterpret_cast<const uint4*>(Arow[i] + k0 + koff);
            af[i].u[0] = p[0];          // K = k0+8h .. +7
            af[i].u[1] = p[2];          // K = k0+16+8h .. +7
        }
#pragma unroll
        for (int j = 0; j < 2; ++j) {
            const uint4* p = reinterpret_cast<const uint4*>(Bcol[j] + k0 + koff);
            bf[j].u[0] = p[0];
            bf[j].u[1] = p[2];
        }
        if (k0 + 32 < K) {              // global_prefetch_b8 for the next K tile
            __builtin_prefetch(Arow[0] + k0 + 32 + koff, 0, 0);
            __builtin_prefetch(Bcol[0] + k0 + 32 + koff, 0, 0);
        }
#pragma unroll
        for (int i = 0; i < 4; ++i)
#pragma unroll
            for (int j = 0; j < 2; ++j)
                acc[i][j] = __builtin_amdgcn_wmma_f32_16x16x32_bf16(
                        false, af[i].v, false, bf[j].v,
                        (short)0, acc[i][j], false, false);
    }

    // C/D layout: lane(n=m16, h=kh), VGPR v -> row v + 8h, col n
#pragma unroll
    for (int i = 0; i < 4; ++i)
#pragma unroll
        for (int j = 0; j < 2; ++j) {
            const int r0 = row0 + i * 16 + 8 * kh;
            const int c  = col0 + j * 16 + m16;
#pragma unroll
            for (int v = 0; v < 8; ++v)
                D[(size_t)(r0 + v) * NCOL + c] = acc[i][j][v];
        }
}

// ---------------------------------------------------------------------------
// K5/K8: BatchNorm stats over one channel row -> fused scale/shift
// grid HC, block 256
// ---------------------------------------------------------------------------
__global__ __launch_bounds__(256) void k_bn_stats(
        const float* __restrict__ Hm,        // [HC x NCOL] row-major
        const float* __restrict__ g, const float* __restrict__ bta,
        float* __restrict__ scale, float* __restrict__ shift)
{
    const int ch = blockIdx.x;
    const int tid = threadIdx.x;
    const float* rowp = Hm + (size_t)ch * NCOL;
    float s = 0.f, q = 0.f;
    for (int i = tid; i < NCOL; i += 256) {
        const float v = rowp[i];
        s += v; q += v * v;
    }
    __shared__ float ss[256], sq[256];
    ss[tid] = s; sq[tid] = q;
    __syncthreads();
    for (int o = 128; o > 0; o >>= 1) {
        if (tid < o) { ss[tid] += ss[tid + o]; sq[tid] += sq[tid + o]; }
        __syncthreads();
    }
    if (tid == 0) {
        const float inv = 1.0f / (float)NCOL;
        const float mean = ss[0] * inv;
        const float var  = sq[0] * inv - mean * mean;   // biased (torch BN)
        const float rs   = rsqrtf(var + BN_EPS);
        const float sc   = g[ch] * rs;
        scale[ch] = sc;
        shift[ch] = bta[ch] - mean * sc;
    }
}

// ---------------------------------------------------------------------------
// K6: apply BN1, SE block (256->10 swish 10->256 sigmoid), gate, repack bf16
//     wave-per-column; lane owns 8 channels; wave32 __shfl_xor reduction.
// grid NCOL/8, block 256 (8 waves)
// ---------------------------------------------------------------------------
__global__ __launch_bounds__(256) void k_se_gate(
        const float* __restrict__ H1,        // [HC x NCOL] f32
        const float* __restrict__ scale1, const float* __restrict__ shift1,
        const float* __restrict__ se_rw, const float* __restrict__ se_rb,
        const float* __restrict__ se_ew, const float* __restrict__ se_eb,
        unsigned short* __restrict__ H1g)    // [NCOL][HC] bf16 column-major
{
    __shared__ float rw[SEC * HC];           // 10 KB
    __shared__ float ew[HC * SEC];           // 10 KB
    __shared__ float sc[HC], sh[HC], eb[HC], rb[SEC];
    const int tid = threadIdx.x;
    for (int i = tid; i < SEC * HC; i += 256) { rw[i] = se_rw[i]; ew[i] = se_ew[i]; }
    sc[tid] = scale1[tid]; sh[tid] = shift1[tid]; eb[tid] = se_eb[tid];
    if (tid < SEC) rb[tid] = se_rb[tid];
    __syncthreads();

    const int lane = tid & 31;
    const int wid  = tid >> 5;
    const size_t col = (size_t)blockIdx.x * 8 + wid;
    const int cbase = lane * 8;

    float hn[8];
    float sacc[SEC];
#pragma unroll
    for (int r = 0; r < SEC; ++r) sacc[r] = 0.f;
#pragma unroll
    for (int i = 0; i < 8; ++i) {
        const int c = cbase + i;
        const float v = H1[(size_t)c * NCOL + col] * sc[c] + sh[c];
        hn[i] = v;
#pragma unroll
        for (int r = 0; r < SEC; ++r) sacc[r] += rw[r * HC + c] * v;
    }
#pragma unroll
    for (int r = 0; r < SEC; ++r) {          // full-wave butterfly sum (wave32)
        float s = sacc[r];
        for (int o = 16; o > 0; o >>= 1) s += __shfl_xor(s, o, 32);
        s += rb[r];
        sacc[r] = s * dev_sigmoid(s);        // swish
    }
    union { unsigned short s[8]; uint4 u; } ou;
#pragma unroll
    for (int i = 0; i < 8; ++i) {
        const int c = cbase + i;
        float e = eb[c];
#pragma unroll
        for (int r = 0; r < SEC; ++r) e += ew[c * SEC + r] * sacc[r];
        ou.s[i] = f32_to_bf16(dev_sigmoid(e) * hn[i]);
    }
    *reinterpret_cast<uint4*>(H1g + col * HC + cbase) = ou.u;   // 16B per lane
}

// ---------------------------------------------------------------------------
// K9: BN2 + ReLU + remap [HC x (B*N)] -> [B, HC, N]
// grid (HC*NCOL)/256, block 256
// ---------------------------------------------------------------------------
__global__ __launch_bounds__(256) void k_bn_relu_out(
        const float* __restrict__ H2,
        const float* __restrict__ scale2, const float* __restrict__ shift2,
        float* __restrict__ out)
{
    const size_t t = (size_t)blockIdx.x * 256 + threadIdx.x;   // over HC*NCOL
    const int ch  = (int)(t >> 16);       // / NCOL
    const int col = (int)(t & 65535);
    const int b = col >> 12;
    const int n = col & 4095;
    const float v = H2[t] * scale2[ch] + shift2[ch];
    out[(((size_t)b * HC + ch) << 12) + n] = fmaxf(v, 0.f);
}

// ---------------------------------------------------------------------------
// Host-side launch
// ---------------------------------------------------------------------------
extern "C" void kernel_launch(void* const* d_in, const int* in_sizes, int n_in,
                              void* d_out, int out_size, void* d_ws, size_t ws_size,
                              hipStream_t stream) {
    const float* unknown = (const float*)d_in[0];
    const float* known   = (const float*)d_in[1];
    const float* uf      = (const float*)d_in[2];
    const float* kf      = (const float*)d_in[3];
    const float* W1      = (const float*)d_in[4];
    const float* g1      = (const float*)d_in[5];
    const float* b1      = (const float*)d_in[6];
    const float* se_rw   = (const float*)d_in[7];
    const float* se_rb   = (const float*)d_in[8];
    const float* se_ew   = (const float*)d_in[9];
    const float* se_eb   = (const float*)d_in[10];
    const float* W2      = (const float*)d_in[11];
    const float* g2      = (const float*)d_in[12];
    const float* b2      = (const float*)d_in[13];
    float* out = (float*)d_out;

    // ---- workspace layout (bytes) ----
    char* ws = (char*)d_ws;
    size_t off = 0;
    int*            idx3   = (int*)           (ws + off); off += (size_t)NCOL * 3 * 4;       // 768 KB
    float*          w3     = (float*)         (ws + off); off += (size_t)NCOL * 3 * 4;       // 768 KB
    unsigned short* Wb1    = (unsigned short*)(ws + off); off += (size_t)HC * CIN * 2;       // 384 KB
    unsigned short* Wb2    = (unsigned short*)(ws + off); off += (size_t)HC * HC * 2;        // 128 KB
    float*          scale1 = (float*)         (ws + off); off += HC * 4;
    float*          shift1 = (float*)         (ws + off); off += HC * 4;
    float*          scale2 = (float*)         (ws + off); off += HC * 4;
    float*          shift2 = (float*)         (ws + off); off += HC * 4;
    off = (off + 255) & ~(size_t)255;
    unsigned short* X      = (unsigned short*)(ws + off);
    float*          H2     = (float*)         (ws + off);                                    // aliases X (X dead after GEMM1)
    off += (size_t)NCOL * CIN * 2;                                                            // 100.7 MB
    float*          H1     = (float*)         (ws + off); off += (size_t)HC * NCOL * 4;      // 67 MB
    unsigned short* H1g    = (unsigned short*)(ws + off); off += (size_t)HC * NCOL * 2;      // 33.5 MB

    // 1) weight packing (f32 -> bf16)
    k_pack_weights<<<(HC * CIN + 255) / 256, 256, 0, stream>>>(W1, W2, Wb1, Wb2);
    // 2) three_nn
    k_three_nn<<<dim3(PN / 128, PB), 128, 0, stream>>>(unknown, known, idx3, w3);
    // 3) interpolate + concat -> X (bf16 col-major)
    k_interp_pack<<<NCOL / 8, 256, 0, stream>>>(uf, kf, idx3, w3, X);
    // 4) conv1 GEMM (WMMA bf16): H1 = W1 * X
    k_gemm_bf16<<<NCOL / 64, 256, 0, stream>>>(Wb1, X, H1, CIN);
    // 5) BN1 stats
    k_bn_stats<<<HC, 256, 0, stream>>>(H1, g1, b1, scale1, shift1);
    // 6) BN1 apply + SE + gate -> H1g (bf16 col-major)
    k_se_gate<<<NCOL / 8, 256, 0, stream>>>(H1, scale1, shift1,
                                            se_rw, se_rb, se_ew, se_eb, H1g);
    // 7) conv2 GEMM (WMMA bf16): H2 = W2 * H1g   (H2 aliases X)
    k_gemm_bf16<<<NCOL / 64, 256, 0, stream>>>(Wb2, H1g, H2, HC);
    // 8) BN2 stats
    k_bn_stats<<<HC, 256, 0, stream>>>(H2, g2, b2, scale2, shift2);
    // 9) BN2 + ReLU + remap to [B, H, N]
    k_bn_relu_out<<<(size_t)HC * NCOL / 256, 256, 0, stream>>>(H2, scale2, shift2, out);
}
